// APPNPXLayerWithGCN_45792941310038
// MI455X (gfx1250) — matrix operand order repaired and checked by reading, not compiled, not run on hardware
//
#include <hip/hip_runtime.h>
#include <hip/hip_bf16.h>
#include <stdint.h>

// ---------- types ----------
typedef __attribute__((ext_vector_type(16))) __bf16   v16bf;
typedef __attribute__((ext_vector_type(8)))  float    v8f;
typedef __attribute__((ext_vector_type(4)))  uint32_t u32x4;
typedef __attribute__((ext_vector_type(4)))  float    f32x4;

#define DIMF 256        // feature dim (D == H == 256)
#define LDSP 264        // padded LDS pitch in bf16 elements (528B row -> conflict-free b128)
#define TILE_M 128      // output rows per block
#define GEMM_LDS ((256 * LDSP + TILE_M * LDSP) * 2)   // W (bf16, transposed) + A tile (bf16) ~198KB

struct alignas(32) BFrag { u32x4 lo, hi; };

static __device__ __forceinline__ v16bf load_frag(const __bf16* p0, const __bf16* p1) {
    BFrag f;
    f.lo = *(const u32x4*)p0;   // 8 bf16 (ds_load_b128)
    f.hi = *(const u32x4*)p1;   // 8 bf16 (ds_load_b128)
    return __builtin_bit_cast(v16bf, f);
}

// ---------- tiny elementwise kernels ----------
__global__ void k_fill1(float* __restrict__ p, int n) {
    int i = blockIdx.x * blockDim.x + threadIdx.x;
    if (i < n) p[i] = 1.0f;   // self-loop contribution to in-degree
}

__global__ void k_deg(const long long* __restrict__ dst, float* __restrict__ deg, int E) {
    int i = blockIdx.x * blockDim.x + threadIdx.x;
    if (i < E) unsafeAtomicAdd(&deg[(int)dst[i]], 1.0f);
}

__global__ void k_rsqrt(float* __restrict__ p, int n) {
    int i = blockIdx.x * blockDim.x + threadIdx.x;
    if (i < n) { float d = p[i]; p[i] = (d > 0.0f) ? rsqrtf(d) : 0.0f; }
}

__global__ void k_zero(float* __restrict__ p, int n) {
    int i = blockIdx.x * blockDim.x + threadIdx.x;
    if (i < n) p[i] = 0.0f;
}

__global__ void k_bias_relu(float* __restrict__ h, const float* __restrict__ b, int total) {
    int i = blockIdx.x * blockDim.x + threadIdx.x;
    if (i < total) {
        float v = h[i] + b[i & (DIMF - 1)];
        h[i] = v > 0.0f ? v : 0.0f;
    }
}

__global__ void k_combine(float* __restrict__ h, const float* __restrict__ prop,
                          const float* __restrict__ h0, int total) {
    int i = blockIdx.x * blockDim.x + threadIdx.x;
    if (i < total) h[i] = 0.9f * prop[i] + 0.1f * h0[i];
}

// ---------- WMMA bf16 GEMM: C[N x 256] = A[N x 256] @ W[256 x 256] ----------
// One block = 128 output rows (amortizes W staging 8x vs a 16-row tile).
// W lives bf16-transposed+padded in LDS (132KB) beside a 128x256 bf16 A tile
// (66KB) -- ~198KB, legal only because CDNA5 gives 320KB LDS per WGP.
// 8 waves each own a 16-row band x 256 cols: 16 f32 accumulator tiles
// (128 VGPRs; wave32 budget is 1024) and 128 v_wmma_f32_16x16x32_bf16 per wave,
// with each A fragment reused across 16 WMMAs.
__global__ void __launch_bounds__(256)
k_gemm256(const float* __restrict__ A, const float* __restrict__ W,
          float* __restrict__ C, int N) {
    extern __shared__ __bf16 lds[];
    __bf16* ldsW = lds;                    // [n][k] transposed, pitch LDSP
    __bf16* ldsA = lds + 256 * LDSP;       // [r][k], pitch LDSP, r in [0,128)

    const int t = threadIdx.x;
    const int rowBase = blockIdx.x * TILE_M;

    // Stage W (transpose to col-major so B fragments are contiguous in K)
    for (int idx = t; idx < 256 * 256; idx += 256) {
        int k = idx >> 8, n = idx & 255;
        ldsW[n * LDSP + k] = (__bf16)W[idx];
    }
    // Stage 128-row A tile (f32 -> bf16)
    for (int idx = t; idx < TILE_M * 256; idx += 256) {
        int r = idx >> 8, k = idx & 255;
        int gr = rowBase + r;
        ldsA[r * LDSP + k] = (gr < N) ? (__bf16)A[(size_t)gr * DIMF + k] : (__bf16)0.0f;
    }
    __syncthreads();

    const int wave = t >> 5;               // wave owns rows [16*wave, 16*wave+16)
    const int lane = t & 31;
    const int l15  = lane & 15;
    const int hi   = (lane >= 16);
    const int rBand = wave * 16;

    v8f acc[16] = {};

    #pragma unroll
    for (int kk = 0; kk < 256; kk += 32) {
        // A fragment (16x32 bf16, ISA 7.12.2): lane row M = lane%16,
        // K = kb..kb+7 and kb+16..kb+23 with kb = kk + (lane>=16 ? 8 : 0)
        const int kbA = kk + (hi ? 8 : 0);
        const __bf16* pa = &ldsA[(rBand + l15) * LDSP + kbA];
        v16bf av = load_frag(pa, pa + 16);
        // B fragments (32x16 bf16): lane col N = lane%16,
        // K = kb..kb+7 and kb+16..kb+23 with kb = kk + (lane>=16 ? 8 : 0)
        const int kbB = kk + (hi ? 8 : 0);
        #pragma unroll
        for (int g = 0; g < 16; ++g) {
            const __bf16* pb = &ldsW[(g * 16 + l15) * LDSP + kbB];
            v16bf bv = load_frag(pb, pb + 16);
            acc[g] = __builtin_amdgcn_wmma_f32_16x16x32_bf16(false, av, false, bv,
                                                             (short)0, acc[g], false, false);
        }
    }

    // C/D layout: lanes 0-15 -> M = r, lanes 16-31 -> M = 8 + r ; N = lane%16
    const int mBase = rowBase + rBand + (hi ? 8 : 0);
    #pragma unroll
    for (int g = 0; g < 16; ++g) {
        #pragma unroll
        for (int r = 0; r < 8; ++r) {
            int m = mBase + r;
            if (m < N) C[(size_t)m * DIMF + g * 16 + l15] = acc[g][r];
        }
    }
}

// ---------- propagation: out[dst] += h[src] * dinv[src]*dinv[dst]  (+ self loops) ----------
// One wave per edge; each lane moves 8 floats. Feature matrix (102MB) is L2-resident
// on MI455X (192MB L2), so the random gather + atomic scatter never leave L2.
__global__ void __launch_bounds__(256)
k_prop(const float* __restrict__ h, const long long* __restrict__ src,
       const long long* __restrict__ dst, const float* __restrict__ dinv,
       float* __restrict__ out, int E, int N) {
    const int wid  = (blockIdx.x * blockDim.x + threadIdx.x) >> 5;
    const int lane = threadIdx.x & 31;
    const int total = E + N;
    if (wid >= total) return;

    int s, d;
    if (wid < E) { s = (int)src[wid]; d = (int)dst[wid]; }
    else         { s = d = wid - E; }                      // self loop
    const float nrm = dinv[s] * dinv[d];

    const f32x4* hp = (const f32x4*)(h + (size_t)s * DIMF) + lane * 2;
    float* op = out + (size_t)d * DIMF + lane * 8;
    f32x4 a = hp[0];
    f32x4 b = hp[1];
    unsafeAtomicAdd(op + 0, a.x * nrm);
    unsafeAtomicAdd(op + 1, a.y * nrm);
    unsafeAtomicAdd(op + 2, a.z * nrm);
    unsafeAtomicAdd(op + 3, a.w * nrm);
    unsafeAtomicAdd(op + 4, b.x * nrm);
    unsafeAtomicAdd(op + 5, b.y * nrm);
    unsafeAtomicAdd(op + 6, b.z * nrm);
    unsafeAtomicAdd(op + 7, b.w * nrm);
}

// ---------- final FC (256 -> 40) + log_softmax ----------
__global__ void __launch_bounds__(256)
k_fc_lsm(const float* __restrict__ h, const float* __restrict__ fw,
         const float* __restrict__ fb, float* __restrict__ out, int N) {
    __shared__ float ldsW[256 * 40];
    __shared__ float ldsB[40];
    const int t = threadIdx.x;
    for (int idx = t; idx < 256 * 40; idx += 256) ldsW[idx] = fw[idx];
    if (t < 40) ldsB[t] = fb[t];
    __syncthreads();

    const int node = blockIdx.x * 256 + t;
    if (node >= N) return;

    float acc[40];
    #pragma unroll
    for (int c = 0; c < 40; ++c) acc[c] = ldsB[c];
    for (int k = 0; k < 256; ++k) {
        float hv = h[(size_t)node * DIMF + k];
        #pragma unroll
        for (int c = 0; c < 40; ++c) acc[c] = fmaf(hv, ldsW[k * 40 + c], acc[c]);
    }
    float m = acc[0];
    #pragma unroll
    for (int c = 1; c < 40; ++c) m = fmaxf(m, acc[c]);
    float s = 0.0f;
    #pragma unroll
    for (int c = 0; c < 40; ++c) s += expf(acc[c] - m);
    const float lse = m + logf(s);
    #pragma unroll
    for (int c = 0; c < 40; ++c) out[(size_t)node * 40 + c] = acc[c] - lse;
}

// ---------- host orchestration ----------
extern "C" void kernel_launch(void* const* d_in, const int* in_sizes, int n_in,
                              void* d_out, int out_size, void* d_ws, size_t ws_size,
                              hipStream_t stream) {
    const float*      x    = (const float*)d_in[0];
    const long long*  ei   = (const long long*)d_in[1];
    const float*      w1   = (const float*)d_in[2];
    const float*      b1   = (const float*)d_in[3];
    const float*      w2   = (const float*)d_in[4];
    const float*      b2   = (const float*)d_in[5];
    const float*      w3   = (const float*)d_in[6];
    const float*      b3   = (const float*)d_in[7];
    const float*      fw   = (const float*)d_in[8];
    const float*      fb   = (const float*)d_in[9];

    const int N = in_sizes[0] / DIMF;
    const int E = in_sizes[1] / 2;
    const long long* esrc = ei;
    const long long* edst = ei + E;

    // workspace carving
    char* ws = (char*)d_ws;
    float* dinv = (float*)ws;
    size_t off  = ((size_t)N * sizeof(float) + 255) & ~(size_t)255;
    const size_t NF = (size_t)N * DIMF;
    float* bufA = (float*)(ws + off);   // GEMM output / APPNP prop scratch
    float* bufB = bufA + NF;            // current features h
    float* h0   = bufB + NF;            // APPNP teleport anchor

    const int totalF   = (int)NF;
    const int ewBlocks = (E + N + 7) / 8;                 // 1 wave per (edge|self-loop)
    const dim3 B256(256);
    const dim3 gF((totalF + 255) / 256);
    const dim3 gN((N + 255) / 256);
    const dim3 gE((E + 255) / 256);
    const dim3 gGemm((N + TILE_M - 1) / TILE_M);
    const dim3 gProp(ewBlocks);

    (void)hipFuncSetAttribute((const void*)k_gemm256,
                              hipFuncAttributeMaxDynamicSharedMemorySize, GEMM_LDS);

    // symmetric GCN normalization
    k_fill1<<<gN, B256, 0, stream>>>(dinv, N);
    k_deg  <<<gE, B256, 0, stream>>>(edst, dinv, E);
    k_rsqrt<<<gN, B256, 0, stream>>>(dinv, N);

    // ---- GCN layer 1 ----
    k_gemm256  <<<gGemm, B256, GEMM_LDS, stream>>>(x, w1, bufA, N);
    k_zero     <<<gF, B256, 0, stream>>>(bufB, totalF);
    k_prop     <<<gProp, B256, 0, stream>>>(bufA, esrc, edst, dinv, bufB, E, N);
    k_bias_relu<<<gF, B256, 0, stream>>>(bufB, b1, totalF);

    // ---- GCN layer 2 ----
    k_gemm256  <<<gGemm, B256, GEMM_LDS, stream>>>(bufB, w2, bufA, N);
    k_zero     <<<gF, B256, 0, stream>>>(bufB, totalF);
    k_prop     <<<gProp, B256, 0, stream>>>(bufA, esrc, edst, dinv, bufB, E, N);
    k_bias_relu<<<gF, B256, 0, stream>>>(bufB, b2, totalF);

    // ---- GCN layer 3 ----
    k_gemm256  <<<gGemm, B256, GEMM_LDS, stream>>>(bufB, w3, bufA, N);
    k_zero     <<<gF, B256, 0, stream>>>(bufB, totalF);
    k_prop     <<<gProp, B256, 0, stream>>>(bufA, esrc, edst, dinv, bufB, E, N);
    k_bias_relu<<<gF, B256, 0, stream>>>(bufB, b3, totalF);

    // ---- APPNP: h <- 0.9 * A_hat h + 0.1 * h0, 10 steps ----
    hipMemcpyAsync(h0, bufB, NF * sizeof(float), hipMemcpyDeviceToDevice, stream);
    for (int it = 0; it < 10; ++it) {
        k_zero   <<<gF, B256, 0, stream>>>(bufA, totalF);
        k_prop   <<<gProp, B256, 0, stream>>>(bufB, esrc, edst, dinv, bufA, E, N);
        k_combine<<<gF, B256, 0, stream>>>(bufB, bufA, h0, totalF);
    }

    // ---- FC + log_softmax ----
    k_fc_lsm<<<gN, B256, 0, stream>>>(bufB, fw, fb, (float*)d_out, N);
}
